// TemporalSelection_37306085933610
// MI455X (gfx1250) — matrix operand order repaired and checked by reading, not compiled, not run on hardware
//
#include <hip/hip_runtime.h>
#include <hip/hip_bf16.h>

// ---------------------------------------------------------------------------
// TemporalSelection for MI455X (gfx1250, wave32, WMMA).
//
// Phases:
//   1) f32->f16 convert of query / w_k / key / w_q         (prep, ~3 MB)
//   2) Kproj[512,512] = query @ w_k^T + b_k                (WMMA f16->f32)
//      Qproj[16,512]  = (key @ w_q^T + b_q) * Dh^-0.5      (WMMA f16->f32)
//   3) per-batch scores/softmax/head-mean/top-8 -> frame idx
//   4) 206 MB float4 gather (bandwidth bound: ~9us @ 23.3 TB/s)
// ---------------------------------------------------------------------------

typedef __attribute__((ext_vector_type(16))) _Float16 v16h;
typedef __attribute__((ext_vector_type(8)))  _Float16 v8h;
typedef __attribute__((ext_vector_type(8)))  float    v8f;

static constexpr int BATCH = 16;
static constexpr int T     = 32;
static constexpr int C     = 512;
static constexpr int H     = 4;
static constexpr int DH    = 128;   // C / H
static constexpr int TOPK  = 8;
static constexpr int TV    = 64;
static constexpr int NPATCH= 196;
static constexpr int CLIP  = 2;     // TV / T

// ---------------------------------------------------------------------------
// f32 -> f16 convert
// ---------------------------------------------------------------------------
__global__ __launch_bounds__(256)
void ts_f32_to_f16(const float* __restrict__ src, _Float16* __restrict__ dst, int n)
{
    int i = blockIdx.x * 256 + threadIdx.x;
    if (i < n) dst[i] = (_Float16)src[i];
}

// ---------------------------------------------------------------------------
// D[M,N] = (A[M,K] @ W[N,K]^T + bias[N]) * scale
// A, W row-major f16 (ld = K); one 16x16 output tile per wave, 8 waves/block.
//
// Fragment layouts per CDNA5 ISA 7.12.2 (wave32):
//   A (16-bit, 16x32): lane m = lane&15, h = lane>>4;
//     halves 0..7  = A[m][k0 + h*8 + 0..7]        (16B load)
//     halves 8..15 = A[m][k0 + 16 + h*8 + 0..7]   (16B load)
//   B (32x16): lane n = lane&15, h = lane>>4;
//     halves 0..7  = W[n][k0 + h*8 + 0..7], halves 8..15 = W[n][k0 + 16 + h*8 + 0..7] (16B loads)
//   C/D (f32 16x16): VGPR i -> row (i + h*8), col (lane&15)
// ---------------------------------------------------------------------------
__global__ __launch_bounds__(256)
void ts_wmma_gemm_nt(const _Float16* __restrict__ A,
                     const _Float16* __restrict__ W,
                     const float*    __restrict__ bias,
                     float*          __restrict__ D,
                     int M, int N, int K, float scale)
{
    const int wave = threadIdx.x >> 5;
    const int lane = threadIdx.x & 31;
    const int tilesN = N >> 4;
    const int tileId = blockIdx.x * 8 + wave;
    const int tm = tileId / tilesN;          // wave-uniform
    const int tn = tileId % tilesN;
    if (tm * 16 >= M) return;                // wave-uniform exit (EXEC stays full)

    const int l16 = lane & 15;
    const int h   = lane >> 4;

    const _Float16* aRow = A + (size_t)(tm * 16 + l16) * K;
    const _Float16* wRow = W + (size_t)(tn * 16 + l16) * K;

    v8f acc = {};
    for (int k0 = 0; k0 < K; k0 += 32) {
        v8h alo = *(const v8h*)(aRow + k0 + h * 8);
        v8h ahi = *(const v8h*)(aRow + k0 + 16 + h * 8);
        v16h a;
#pragma unroll
        for (int i = 0; i < 8; ++i) { a[i] = alo[i]; a[i + 8] = ahi[i]; }
        v8h blo = *(const v8h*)(wRow + k0 + h * 8);
        v8h bhi = *(const v8h*)(wRow + k0 + 16 + h * 8);
        v16h b;
#pragma unroll
        for (int i = 0; i < 8; ++i) { b[i] = blo[i]; b[i + 8] = bhi[i]; }
        // (neg_a, A, neg_b, B, c_mod, C, reuse_a, reuse_b)
        acc = __builtin_amdgcn_wmma_f32_16x16x32_f16(
            false, a, false, b, (short)0, acc, false, false);
    }

    const float bn = bias[tn * 16 + l16];
    float* dBase = D + (size_t)(tm * 16 + h * 8) * N + tn * 16 + l16;
#pragma unroll
    for (int i = 0; i < 8; ++i)
        dBase[(size_t)i * N] = (acc[i] + bn) * scale;
}

// ---------------------------------------------------------------------------
// Per batch: scores[h][t] = q[b,h,:] . k[b,t,h,:]; softmax over t; mean over h;
// top-8 (argsort-last-k semantics), sorted ascending; expand by clip_len=2.
// Block = 128 threads = 4 waves; wave == head, lane == t.
// ---------------------------------------------------------------------------
__global__ __launch_bounds__(128)
void ts_scores_topk(const float* __restrict__ Qp,   // [BATCH, C] (already scaled)
                    const float* __restrict__ Kp,   // [BATCH*T, C]
                    int* __restrict__ fidx)         // [BATCH, 2*TOPK]
{
    const int b = blockIdx.x;
    const int h = threadIdx.x >> 5;
    const int t = threadIdx.x & 31;

    const float* q = Qp + (size_t)b * C + h * DH;
    const float* k = Kp + (size_t)(b * T + t) * C + h * DH;
    float s = 0.f;
#pragma unroll 4
    for (int d = 0; d < DH; ++d) s += q[d] * k[d];

    // softmax across the 32 lanes (one head per wave)
    float m = s;
#pragma unroll
    for (int o = 16; o; o >>= 1) m = fmaxf(m, __shfl_xor(m, o, 32));
    float e = __expf(s - m);
    float sum = e;
#pragma unroll
    for (int o = 16; o; o >>= 1) sum += __shfl_xor(sum, o, 32);
    const float attn = e / sum;

    __shared__ float sm[H][T];
    sm[h][t] = attn;
    __syncthreads();

    if (h == 0) {
        float v = 0.25f * (sm[0][t] + sm[1][t] + sm[2][t] + sm[3][t]);

        int sel[TOPK];
#pragma unroll
        for (int r = 0; r < TOPK; ++r) {
            float mv = v;
            int   mi = t;
#pragma unroll
            for (int o = 16; o; o >>= 1) {
                float ov = __shfl_xor(mv, o, 32);
                int   oi = __shfl_xor(mi, o, 32);
                // tie -> larger index (matches argsort[..., -k:] on stable sort)
                if (ov > mv || (ov == mv && oi > mi)) { mv = ov; mi = oi; }
            }
            sel[r] = mi;                       // all lanes agree
            if (t == mi) v = -__builtin_inff();
        }

        if (t == 0) {
            // sort the 8 winning indices ascending
#pragma unroll
            for (int i = 1; i < TOPK; ++i) {
                int key = sel[i], j = i - 1;
                while (j >= 0 && sel[j] > key) { sel[j + 1] = sel[j]; --j; }
                sel[j + 1] = key;
            }
#pragma unroll
            for (int j = 0; j < TOPK; ++j) {
                fidx[b * (TOPK * CLIP) + 2 * j    ] = sel[j] * CLIP;
                fidx[b * (TOPK * CLIP) + 2 * j + 1] = sel[j] * CLIP + 1;
            }
        }
    }
}

// ---------------------------------------------------------------------------
// out[b, j, :, :] = value[b, fidx[b*16+j], :, :]
// One (b,j) slice per blockIdx.y; 196*512 f32 = 25088 float4 = 98 blocks x 256.
// Pure b128 streaming copy -> HBM bound.
// ---------------------------------------------------------------------------
__global__ __launch_bounds__(256)
void ts_gather_frames(const float4* __restrict__ value,  // [BATCH*TV*N*C/4]
                      const int*    __restrict__ fidx,   // [BATCH*16]
                      float4*       __restrict__ out)    // [BATCH*16*N*C/4]
{
    const int slice = blockIdx.y;            // b*16 + j
    const int b = slice >> 4;
    const int f = fidx[slice];
    const size_t SLICE4 = (size_t)NPATCH * C / 4;   // 25088
    const float4* src = value + ((size_t)b * TV + f) * SLICE4;
    float4*       dst = out   + (size_t)slice * SLICE4;
    const int i = blockIdx.x * 256 + threadIdx.x;
    dst[i] = src[i];
}

// ---------------------------------------------------------------------------
extern "C" void kernel_launch(void* const* d_in, const int* in_sizes, int n_in,
                              void* d_out, int out_size, void* d_ws, size_t ws_size,
                              hipStream_t stream)
{
    const float* query = (const float*)d_in[0];   // [B,T,C]
    const float* key   = (const float*)d_in[1];   // [B,C]
    const float* value = (const float*)d_in[2];   // [B,TV,N,C]
    const float* w_q   = (const float*)d_in[3];   // [C,C]
    const float* w_k   = (const float*)d_in[4];   // [C,C]
    const float* b_q   = (const float*)d_in[5];   // [C]
    const float* b_k   = (const float*)d_in[6];   // [C]
    float* out = (float*)d_out;

    // workspace layout (256B aligned chunks)
    char* ws = (char*)d_ws;
    size_t off = 0;
    auto alloc = [&](size_t bytes) { char* p = ws + off; off = (off + bytes + 255) & ~(size_t)255; return p; };

    _Float16* qf16  = (_Float16*)alloc((size_t)BATCH * T * C * 2);  // query f16
    _Float16* wkf16 = (_Float16*)alloc((size_t)C * C * 2);
    _Float16* kyf16 = (_Float16*)alloc((size_t)BATCH * C * 2);      // key f16
    _Float16* wqf16 = (_Float16*)alloc((size_t)C * C * 2);
    float*    Kproj = (float*)alloc((size_t)BATCH * T * C * 4);     // [512,512]
    float*    Qproj = (float*)alloc((size_t)BATCH * C * 4);         // [16,512]
    int*      fidx  = (int*)alloc((size_t)BATCH * TOPK * CLIP * 4); // [16,16]

    // 1) converts
    {
        int n = BATCH * T * C;  // 262144
        ts_f32_to_f16<<<(n + 255) / 256, 256, 0, stream>>>(query, qf16, n);
        n = C * C;
        ts_f32_to_f16<<<(n + 255) / 256, 256, 0, stream>>>(w_k, wkf16, n);
        ts_f32_to_f16<<<(n + 255) / 256, 256, 0, stream>>>(w_q, wqf16, n);
        n = BATCH * C;
        ts_f32_to_f16<<<(n + 255) / 256, 256, 0, stream>>>(key, kyf16, n);
    }

    // 2) projections via WMMA
    //    Kproj: M=512, N=512, K=512 -> 1024 tiles -> 128 blocks of 8 waves
    ts_wmma_gemm_nt<<<128, 256, 0, stream>>>(qf16, wkf16, b_k, Kproj,
                                             BATCH * T, C, C, 1.0f);
    //    Qproj: M=16, N=512 -> 32 tiles -> 4 blocks; scale = Dh^-0.5
    const float dh_rsqrt = 0.08838834764831845f;  // 1/sqrt(128)
    ts_wmma_gemm_nt<<<4, 256, 0, stream>>>(kyf16, wqf16, b_q, Qproj,
                                           BATCH, C, C, dh_rsqrt);

    // 3) scores + softmax + head mean + top-k -> frame indices
    ts_scores_topk<<<BATCH, 128, 0, stream>>>(Qproj, Kproj, fidx);

    // 4) gather (bandwidth-bound phase)
    dim3 ggrid(98, BATCH * TOPK * CLIP);  // 98*256 = 25088 float4 per slice
    ts_gather_frames<<<ggrid, 256, 0, stream>>>((const float4*)value, fidx,
                                                (float4*)out);
}
